// MultiheadGQA_v2_9740985828012
// MI455X (gfx1250) — matrix operand order repaired and checked by reading, not compiled, not run on hardware
//
#include <hip/hip_runtime.h>
#include <hip/hip_bf16.h>

// ---------------------------------------------------------------------------
// Problem constants (B=2, N=2048, E=2048, HQ=16, HK=4, D=128, KVE=512)
// ---------------------------------------------------------------------------
constexpr int BQ   = 2;
constexpr int NSEQ = 2048;
constexpr int EMB  = 2048;
constexpr int HQn  = 16;
constexpr int HKn  = 4;
constexpr int DH   = 128;
constexpr int KVE_ = 512;

typedef __attribute__((ext_vector_type(16))) __bf16 v16bf;
typedef __attribute__((ext_vector_type(8)))  float  v8f;

// ---------------------------------------------------------------------------
// WMMA helpers (CDNA5 16x16x32 bf16, f32 accumulate)
// ---------------------------------------------------------------------------
__device__ __forceinline__ v8f wmma_bf16(v16bf a, v16bf b, v8f c) {
  return __builtin_amdgcn_wmma_f32_16x16x32_bf16(
      /*neg_a=*/false, a, /*neg_b=*/false, b,
      /*c_mod=*/(short)0, c, /*reuse_a=*/false, /*reuse_b=*/false);
}

// A-matrix 16x32 bf16 fragment.  base must be wave-uniform; all lane
// variation lives in a single 32-bit index so the backend can use
// SADDR + 32-bit VOFFSET addressing (cheap, few VGPRs).
// Layout: lane L -> row M = L%16; lanes 0-15 hold K=0..15,
// lanes 16-31 hold K=16..31.
__device__ __forceinline__ v16bf load_a_frag(const __bf16* base, int ld,
                                             int r0, int k0) {
  const int lane = threadIdx.x & 31;
  const int idx = (r0 + (lane & 15)) * ld + k0 + (lane >> 4) * 16;
  v16bf f;
#pragma unroll
  for (int i = 0; i < 16; ++i) f[i] = base[idx + i];
  return f;
}

// B-matrix 32x16 bf16 fragment from a K-contiguous matrix (B = W^T, W
// row-major (N,K)).  Lane L -> column n = L%16; lanes 0-15 hold
// K=k0..k0+15, lanes 16-31 hold K=k0+16..k0+31 (contiguous 32B per lane).
__device__ __forceinline__ v16bf load_bT_frag(const __bf16* base, int ldk,
                                              int n0, int k0) {
  const int lane = threadIdx.x & 31;
  const int idx = (n0 + (lane & 15)) * ldk + k0 + (lane >> 4) * 16;
  v16bf f;
#pragma unroll
  for (int i = 0; i < 16; ++i) f[i] = base[idx + i];
  return f;
}

// ---------------------------------------------------------------------------
// f32 -> bf16 cast
// ---------------------------------------------------------------------------
__global__ void cvt_f32_bf16(const float* __restrict__ x,
                             __bf16* __restrict__ y, int n) {
  for (int i = blockIdx.x * blockDim.x + threadIdx.x; i < n;
       i += gridDim.x * blockDim.x)
    y[i] = (__bf16)x[i];
}

// ---------------------------------------------------------------------------
// Sinusoidal positional encoding value pe[pos][e] for model dim dmodel.
// ---------------------------------------------------------------------------
__device__ __forceinline__ float sin_pe(int pos, int e, int dmodel) {
  const int i2 = e & ~1;
  const float ang =
      (float)pos * __powf(10000.f, -(float)i2 / (float)dmodel);
  return (e & 1) ? __cosf(ang) : __sinf(ang);
}

// ---------------------------------------------------------------------------
// GEMM: acc[M,N] = A[M,K](bf16) @ W[N,K]^T(bf16) + bias[N]
// 256 threads = 8 waves (4x2), wave tile 64x32, block tile 256x64.
// Epilogue MODE:
//   0: Cf[row*Nn+col] = acc + bias                        (f32 out)
//   1: q-pack: +PE(EMB), *1/D, bf16 -> (b,h,n,d)
//   2: k-pack: +PE(KVE),        bf16 -> (b,hk,n,d)
//   3: v-pack:                  bf16 -> (b,hk,d,n)  (transposed for PV frags)
// ---------------------------------------------------------------------------
template <int MODE>
__global__ __launch_bounds__(256) void gemm_bf16(
    const __bf16* __restrict__ A, const __bf16* __restrict__ W,
    const float* __restrict__ bias, float* __restrict__ Cf,
    __bf16* __restrict__ Cb, int Nn, int K) {
  const int wave = __builtin_amdgcn_readfirstlane(threadIdx.x >> 5);
  const int wr = wave >> 1, wc = wave & 1;
  const int m0 = blockIdx.y * 256 + wr * 64;
  const int n0 = blockIdx.x * 64 + wc * 32;

  v8f acc[4][2];
#pragma unroll
  for (int i = 0; i < 4; ++i)
#pragma unroll
    for (int j = 0; j < 2; ++j)
#pragma unroll
      for (int e = 0; e < 8; ++e) acc[i][j][e] = 0.f;

  for (int k0 = 0; k0 < K; k0 += 32) {
    __builtin_prefetch(A + (size_t)m0 * K + k0 + 512, 0, 0);  // stream ahead
    v16bf a[4];
#pragma unroll
    for (int mi = 0; mi < 4; ++mi)
      a[mi] = load_a_frag(A, K, m0 + mi * 16, k0);
    v16bf b0 = load_bT_frag(W, K, n0, k0);
    v16bf b1 = load_bT_frag(W, K, n0 + 16, k0);
#pragma unroll
    for (int mi = 0; mi < 4; ++mi) {
      acc[mi][0] = wmma_bf16(a[mi], b0, acc[mi][0]);
      acc[mi][1] = wmma_bf16(a[mi], b1, acc[mi][1]);
    }
  }

  const int lane = threadIdx.x & 31;
  const int half = lane >> 4;
#pragma unroll
  for (int mi = 0; mi < 4; ++mi)
#pragma unroll
    for (int ni = 0; ni < 2; ++ni)
#pragma unroll
      for (int r = 0; r < 8; ++r) {
        const int row = m0 + mi * 16 + r + 8 * half;   // global row (b*N+n)
        const int col = n0 + ni * 16 + (lane & 15);    // output feature
        const float v = acc[mi][ni][r] + bias[col];
        if (MODE == 0) {
          Cf[(size_t)row * Nn + col] = v;
        } else {
          const int n = row & (NSEQ - 1);
          const int b = row >> 11;  // row / NSEQ
          const int h = col >> 7, d = col & 127;
          if (MODE == 1) {
            const float q = (v + sin_pe(n, col, EMB)) * (1.f / (float)DH);
            Cb[(((size_t)b * HQn + h) * NSEQ + n) * DH + d] = (__bf16)q;
          } else if (MODE == 2) {
            const float kk = v + sin_pe(n, col, KVE_);
            Cb[(((size_t)b * HKn + h) * NSEQ + n) * DH + d] = (__bf16)kk;
          } else {  // MODE == 3, transposed V
            Cb[(((size_t)b * HKn + h) * DH + d) * NSEQ + n] = (__bf16)v;
          }
        }
      }
}

// ---------------------------------------------------------------------------
// Causal flash attention with GQA + relative position bias.
// grid = (N/128, HQ, B), 256 threads = 8 fully independent waves.
// Wave owns 16 q-rows; Bc = 64; per-wave causal tile count.
// K (b,hk,n,d) and V^T (b,hk,d,n) are both K-contiguous -> all WMMA B
// fragments stream straight from global/L2 via SADDR+VOFFSET loads.
// LDS used only for the wave-private P layout conversion (C -> A fragment),
// padded to stride 72 for bank-conflict-free b128 re-reads.
// Out: (b,n,E) f32.
// ---------------------------------------------------------------------------
__global__ __launch_bounds__(256) void flash_attn(
    const __bf16* __restrict__ Qp, const __bf16* __restrict__ Kp,
    const __bf16* __restrict__ Vt, const float* __restrict__ rel_w,
    float* __restrict__ Out) {
  constexpr int BR = 128, BC = 64, PSTR = 72;
  __shared__ __bf16 Ps[8 * 16 * PSTR];  // 18 KB, per-wave 16 x BC (padded)

  const int qt = blockIdx.x, h = blockIdx.y, b = blockIdx.z;
  const int hk = h & (HKn - 1);  // group-major repeat: hk = h % HK
  const int wave = __builtin_amdgcn_readfirstlane(threadIdx.x >> 5);
  const int lane = threadIdx.x & 31;
  const int half = lane >> 4;
  const int i_base = qt * BR + wave * 16;

  const __bf16* Qb = Qp + (((size_t)b * HQn + h) * NSEQ) * DH;
  const __bf16* Kb = Kp + ((size_t)b * HKn + hk) * NSEQ * DH;
  const __bf16* Vb = Vt + ((size_t)b * HKn + hk) * (size_t)DH * NSEQ;

  v16bf qf[4];
#pragma unroll
  for (int kk = 0; kk < 4; ++kk)
    qf[kk] = load_a_frag(Qb, DH, i_base, kk * 32);

  v8f o[8];
#pragma unroll
  for (int nb = 0; nb < 8; ++nb)
#pragma unroll
    for (int e = 0; e < 8; ++e) o[nb][e] = 0.f;
  float mrow[8], lrow[8];
#pragma unroll
  for (int r = 0; r < 8; ++r) { mrow[r] = -3.0e38f; lrow[r] = 0.f; }

  const float wh = rel_w[h];
  const float cj = wh * (1.f / (float)NSEQ);   // rel = wh + (j-i)*cj (j<=i)
  const int ntile = (i_base >> 6) + 1;         // per-wave causal tile count
  __bf16* Pw = Ps + wave * 16 * PSTR;

  for (int t = 0; t < ntile; ++t) {
    const int j0 = t * BC;
    __builtin_prefetch(Kb + (size_t)(j0 + BC) * DH, 0, 0);

    // S = Q K^T : 4 column blocks; batch the 4 K fragments of a block so
    // several b128 loads are outstanding before the WMMAs consume them.
    v8f s[4];
#pragma unroll
    for (int cb = 0; cb < 4; ++cb) {
      v16bf kf[4];
#pragma unroll
      for (int kk = 0; kk < 4; ++kk)
        kf[kk] = load_bT_frag(Kb, DH, j0 + cb * 16, kk * 32);
#pragma unroll
      for (int e = 0; e < 8; ++e) s[cb][e] = 0.f;
#pragma unroll
      for (int kk = 0; kk < 4; ++kk)
        s[cb] = wmma_bf16(qf[kk], kf[kk], s[cb]);
    }

    // Online softmax per row (rows split across lane halves); P -> LDS bf16.
#pragma unroll
    for (int r = 0; r < 8; ++r) {
      const int i = i_base + r + 8 * half;
      const float bias_i = wh - (float)i * cj;
      float pl[4];
      float mx = -3.0e38f;
#pragma unroll
      for (int cb = 0; cb < 4; ++cb) {
        const int j = j0 + cb * 16 + (lane & 15);
        float lg;
        if (j > i) lg = -3.0e38f;                       // causal mask
        else       lg = s[cb][r] + bias_i + (float)j * cj;  // rel bias
        pl[cb] = lg;
        mx = fmaxf(mx, lg);
      }
#pragma unroll
      for (int off = 1; off < 16; off <<= 1)
        mx = fmaxf(mx, __shfl_xor(mx, off, 32));
      const float mnew = fmaxf(mrow[r], mx);
      const float corr = __expf(mrow[r] - mnew);
      mrow[r] = mnew;
      float rs = 0.f;
#pragma unroll
      for (int cb = 0; cb < 4; ++cb) {
        const float pv = __expf(pl[cb] - mnew);
        rs += pv;
        Pw[(r + 8 * half) * PSTR + cb * 16 + (lane & 15)] = (__bf16)pv;
      }
#pragma unroll
      for (int off = 1; off < 16; off <<= 1) rs += __shfl_xor(rs, off, 32);
      lrow[r] = lrow[r] * corr + rs;
#pragma unroll
      for (int nb = 0; nb < 8; ++nb) o[nb][r] *= corr;
    }

    // O += P V.  P re-read from wave-private LDS as A fragments (in-order
    // per-wave DS pipeline, no barrier needed); V^T fragments from global.
    v16bf pf0 = load_a_frag(Pw, PSTR, 0, 0);
    v16bf pf1 = load_a_frag(Pw, PSTR, 0, 32);
#pragma unroll
    for (int nb = 0; nb < 8; ++nb) {
      v16bf vf0 = load_bT_frag(Vb, NSEQ, nb * 16, j0);
      v16bf vf1 = load_bT_frag(Vb, NSEQ, nb * 16, j0 + 32);
      o[nb] = wmma_bf16(pf0, vf0, o[nb]);
      o[nb] = wmma_bf16(pf1, vf1, o[nb]);
    }
  }

  float* Ob = Out + ((size_t)b * NSEQ + i_base) * EMB + h * DH;
#pragma unroll
  for (int nb = 0; nb < 8; ++nb)
#pragma unroll
    for (int r = 0; r < 8; ++r) {
      const int row = r + 8 * half;
      const int col = nb * 16 + (lane & 15);
      Ob[(size_t)row * EMB + col] = o[nb][r] / lrow[r];
    }
}

// ---------------------------------------------------------------------------
// MAGNETO LayerNorm over E, f32 in -> bf16 out. One block per row.
// ---------------------------------------------------------------------------
__global__ __launch_bounds__(256) void layernorm_bf16(
    const float* __restrict__ X, const float* __restrict__ g,
    const float* __restrict__ bta, __bf16* __restrict__ Y) {
  __shared__ float sh1[8], sh2[8];
  const size_t row = blockIdx.x;
  const float* x = X + row * EMB;
  float s = 0.f, s2 = 0.f;
  for (int e = threadIdx.x; e < EMB; e += 256) {
    const float v = x[e];
    s += v;
    s2 += v * v;
  }
#pragma unroll
  for (int off = 1; off < 32; off <<= 1) {
    s += __shfl_xor(s, off, 32);
    s2 += __shfl_xor(s2, off, 32);
  }
  const int wave = threadIdx.x >> 5, lane = threadIdx.x & 31;
  if (lane == 0) { sh1[wave] = s; sh2[wave] = s2; }
  __syncthreads();
  float ts = 0.f, ts2 = 0.f;
#pragma unroll
  for (int w = 0; w < 8; ++w) { ts += sh1[w]; ts2 += sh2[w]; }
  const float mu = ts * (1.f / EMB);
  const float var = ts2 * (1.f / EMB) - mu * mu;
  const float rs = rsqrtf(var + 1e-5f);
  __bf16* y = Y + row * EMB;
  for (int e = threadIdx.x; e < EMB; e += 256)
    y[e] = (__bf16)((x[e] - mu) * rs * g[e] + bta[e]);
}

// ---------------------------------------------------------------------------
// Host side
// ---------------------------------------------------------------------------
extern "C" void kernel_launch(void* const* d_in, const int* in_sizes, int n_in,
                              void* d_out, int out_size, void* d_ws,
                              size_t ws_size, hipStream_t stream) {
  (void)in_sizes; (void)n_in; (void)out_size; (void)ws_size;
  const float* query = (const float*)d_in[0];
  const float* key_  = (const float*)d_in[1];
  const float* value = (const float*)d_in[2];
  const float* Wq = (const float*)d_in[3];
  const float* bq = (const float*)d_in[4];
  const float* Wk = (const float*)d_in[5];
  const float* bk = (const float*)d_in[6];
  const float* Wv = (const float*)d_in[7];
  const float* bv = (const float*)d_in[8];
  const float* Wo = (const float*)d_in[9];
  const float* bo = (const float*)d_in[10];
  const float* ln_g = (const float*)d_in[11];
  const float* ln_b = (const float*)d_in[12];
  const float* rel_w = (const float*)d_in[13];
  float* out = (float*)d_out;

  const size_t nBNE = (size_t)BQ * NSEQ * EMB;   // 8.4M
  const size_t nBNK = (size_t)BQ * NSEQ * KVE_;  // 2.1M
  const size_t nWqq = (size_t)EMB * EMB;
  const size_t nWkv = (size_t)KVE_ * EMB;

  char* ws = (char*)d_ws;
  size_t off = 0;
  auto alloc = [&](size_t bytes) -> void* {
    void* p = ws + off;
    off += (bytes + 255) & ~(size_t)255;
    return p;
  };
  __bf16* qin   = (__bf16*)alloc(nBNE * 2);  // reused later for normed
  __bf16* kin   = (__bf16*)alloc(nBNE * 2);
  __bf16* vin   = (__bf16*)alloc(nBNE * 2);
  __bf16* wq    = (__bf16*)alloc(nWqq * 2);
  __bf16* wk    = (__bf16*)alloc(nWkv * 2);
  __bf16* wv    = (__bf16*)alloc(nWkv * 2);
  __bf16* wo    = (__bf16*)alloc(nWqq * 2);
  __bf16* qpack = (__bf16*)alloc(nBNE * 2);  // (b,h,n,d)
  __bf16* kpack = (__bf16*)alloc(nBNK * 2);  // (b,hk,n,d)
  __bf16* vpckT = (__bf16*)alloc(nBNK * 2);  // (b,hk,d,n)
  float*  attn  = (float*)alloc(nBNE * 4);   // (b,n,E)

  const int CT = 256, CB = 4096;
  cvt_f32_bf16<<<CB, CT, 0, stream>>>(query, qin, (int)nBNE);
  cvt_f32_bf16<<<CB, CT, 0, stream>>>(key_,  kin, (int)nBNE);
  cvt_f32_bf16<<<CB, CT, 0, stream>>>(value, vin, (int)nBNE);
  cvt_f32_bf16<<<CB, CT, 0, stream>>>(Wq, wq, (int)nWqq);
  cvt_f32_bf16<<<CB, CT, 0, stream>>>(Wk, wk, (int)nWkv);
  cvt_f32_bf16<<<CB, CT, 0, stream>>>(Wv, wv, (int)nWkv);
  cvt_f32_bf16<<<CB, CT, 0, stream>>>(Wo, wo, (int)nWqq);

  const int Mrows = BQ * NSEQ;  // 4096
  // Projections with fused bias + PE + head packing in the WMMA epilogue.
  gemm_bf16<1><<<dim3(EMB / 64, Mrows / 256), 256, 0, stream>>>(
      qin, wq, bq, nullptr, qpack, EMB, EMB);
  gemm_bf16<2><<<dim3(KVE_ / 64, Mrows / 256), 256, 0, stream>>>(
      kin, wk, bk, nullptr, kpack, KVE_, EMB);
  gemm_bf16<3><<<dim3(KVE_ / 64, Mrows / 256), 256, 0, stream>>>(
      vin, wv, bv, nullptr, vpckT, KVE_, EMB);

  flash_attn<<<dim3(NSEQ / 128, HQn, BQ), 256, 0, stream>>>(
      qpack, kpack, vpckT, rel_w, attn);

  __bf16* normed = qin;  // reuse: qin no longer needed
  layernorm_bf16<<<Mrows, 256, 0, stream>>>(attn, ln_g, ln_b, normed);

  gemm_bf16<0><<<dim3(EMB / 64, Mrows / 256), 256, 0, stream>>>(
      normed, wo, bo, out, nullptr, EMB, EMB);
}
